// SpectralMoEDictionaryCrossAttention_4715874091100
// MI455X (gfx1250) — matrix operand order repaired and checked by reading, not compiled, not run on hardware
//
#include <hip/hip_runtime.h>
#include <hip/hip_bf16.h>
#include <stdint.h>

// ---------------------------------------------------------------------------
// CDNA5 (gfx1250, wave32) implementation of SpectralMoEDictionaryCrossAttention
// All dense GEMMs go through v_wmma_f32_16x16x32_bf16 (bf16 in, f32 acc),
// with the shared B tile staged through LDS (double-buffered).
// ---------------------------------------------------------------------------

typedef __attribute__((ext_vector_type(16))) __bf16          bf16x16;
typedef __attribute__((ext_vector_type(8)))  __bf16          bf16x8;
typedef __attribute__((ext_vector_type(8)))  float           f32x8;
typedef __attribute__((ext_vector_type(16))) unsigned short  u16x16;
typedef __attribute__((ext_vector_type(8)))  unsigned short  u16x8;

#define DEV __device__ __forceinline__

DEV float gelu_f(float x) { return 0.5f * x * (1.0f + erff(x * 0.70710678118654752f)); }
DEV float sigmoid_f(float x) { return 1.0f / (1.0f + expf(-x)); }

DEV unsigned short bf16_cvt(float f) {
    __bf16 h = (__bf16)f;                       // hardware RNE convert
    return __builtin_bit_cast(unsigned short, h);
}

// ---------------------------------------------------------------------------
// WMMA GEMM:  C[M,N] = epilogue( A[M,K] (f32->bf16) @ Wt[N,K] (bf16) )
// epilogue: +bias[n]; *rowScale[m*rsStride]; (+= C old if accFlag); ACT
// Block: 256 threads = 8 waves. Wave tile: 32 rows x 64 cols (8 WMMA / K-step).
// B tile (64 x 32 bf16) is staged in LDS, double-buffered, padded stride.
// FULLN: N % 64 == 0 -> branch-free inner loop.
// ---------------------------------------------------------------------------
#define LDSB_STRIDE 40   // ushorts per B row in LDS (padding vs bank conflicts)

template <int ACT, bool FULLN> // ACT: 0 = none, 1 = exact GELU
__global__ __launch_bounds__(256)
void gemm_wmma_bf16(const float* __restrict__ A, int ldA,
                    const unsigned short* __restrict__ Wt, int ldW,
                    const float* __restrict__ bias,
                    float* __restrict__ C, int ldC,
                    int M, int N, int K,
                    const float* __restrict__ rowScale, int rsStride,
                    int accFlag)
{
    __shared__ unsigned short ldsB[2][64 * LDSB_STRIDE];

    const int tid  = threadIdx.x;
    const int wave = tid >> 5;
    const int lane = tid & 31;
    const int half = lane >> 4;
    const int l16  = lane & 15;
    const int tileM  = (blockIdx.y * 8 + wave) * 32;
    const int tileN0 = blockIdx.x * 64;
    const bool waveActive = (tileM < M);

    // cooperative staging map: 256 threads x 8 ushorts = 64 rows x 32 k
    const int sN = (tid * 8) >> 5;          // 0..63
    const int sK = (tid * 8) & 31;          // 0,8,16,24
    const bool sValid = FULLN || ((tileN0 + sN) < N);
    const unsigned short* gB = Wt + (size_t)(tileN0 + sN) * ldW + sK;

    const int nIter = K >> 5;

    f32x8 acc[2][4];
#pragma unroll
    for (int rr = 0; rr < 2; ++rr)
#pragma unroll
        for (int j = 0; j < 4; ++j)
#pragma unroll
            for (int r = 0; r < 8; ++r) acc[rr][j][r] = 0.0f;

    // preload B tile 0
    {
        u16x8 v;
#pragma unroll
        for (int i = 0; i < 8; ++i) v[i] = 0;
        if (sValid) v = *(const u16x8*)gB;
        *(u16x8*)&ldsB[0][sN * LDSB_STRIDE + sK] = v;
    }
    __syncthreads();

    for (int it = 0; it < nIter; ++it) {
        const int k0  = it << 5;
        const int cur = it & 1;

        // stage next B tile into the other buffer
        if (it + 1 < nIter) {
            u16x8 v;
#pragma unroll
            for (int i = 0; i < 8; ++i) v[i] = 0;
            if (sValid) v = *(const u16x8*)(gB + k0 + 32);
            *(u16x8*)&ldsB[cur ^ 1][sN * LDSB_STRIDE + sK] = v;
        }

        if (waveActive) {
            // ---- two A fragments (rows tileM+l16, tileM+16+l16) ----
            bf16x16 afrag[2];
#pragma unroll
            for (int rr = 0; rr < 2; ++rr) {
                const float* ap = A + (size_t)(tileM + rr * 16 + l16) * ldA + k0 + half * 8;
                f32x8 a0 = *(const f32x8*)(ap);
                f32x8 a1 = *(const f32x8*)(ap + 16);
                bf16x8 c0 = __builtin_convertvector(a0, bf16x8);
                bf16x8 c1 = __builtin_convertvector(a1, bf16x8);
                afrag[rr] = __builtin_shufflevector(c0, c1,
                    0, 1, 2, 3, 4, 5, 6, 7, 8, 9, 10, 11, 12, 13, 14, 15);
            }
            if (k0 + 64 < K)
                __builtin_prefetch(A + (size_t)(tileM + l16) * ldA + k0 + 64, 0, 1);

#pragma unroll
            for (int j = 0; j < 4; ++j) {
                if (FULLN || (tileN0 + j * 16 < N)) {
                    const unsigned short* lp =
                        &ldsB[cur][(j * 16 + l16) * LDSB_STRIDE + half * 8];
                    bf16x8 b0 = __builtin_bit_cast(bf16x8, *(const u16x8*)(lp));
                    bf16x8 b1 = __builtin_bit_cast(bf16x8, *(const u16x8*)(lp + 16));
                    bf16x16 bfrag = __builtin_shufflevector(b0, b1,
                        0, 1, 2, 3, 4, 5, 6, 7, 8, 9, 10, 11, 12, 13, 14, 15);
                    acc[0][j] = __builtin_amdgcn_wmma_f32_16x16x32_bf16(
                        false, afrag[0], false, bfrag, (short)0, acc[0][j], false, false);
                    acc[1][j] = __builtin_amdgcn_wmma_f32_16x16x32_bf16(
                        false, afrag[1], false, bfrag, (short)0, acc[1][j], false, false);
                }
            }
        }
        __syncthreads();
    }

    if (!waveActive) return;

#pragma unroll
    for (int rr = 0; rr < 2; ++rr) {
#pragma unroll
        for (int j = 0; j < 4; ++j) {
            const int n0 = tileN0 + j * 16;
            if (!FULLN && n0 >= N) continue;
            const int n = n0 + l16;
            const float bv = bias ? bias[n] : 0.0f;
#pragma unroll
            for (int r = 0; r < 8; ++r) {
                const int m = tileM + rr * 16 + half * 8 + r;
                float v = acc[rr][j][r] + bv;
                if (rowScale) v *= rowScale[(size_t)m * rsStride];
                if (accFlag)  v += C[(size_t)m * ldC + n];
                if (ACT == 1) v = gelu_f(v);
                C[(size_t)m * ldC + n] = v;
            }
        }
    }
}

// ---------------------------------------------------------------------------
// Weight packing: f32 [K][N] -> bf16 [N][K]   (and row-major variant)
// ---------------------------------------------------------------------------
__global__ void pack_transpose_k(const float* __restrict__ W,
                                 unsigned short* __restrict__ Wt, int K, int N)
{
    int idx = blockIdx.x * blockDim.x + threadIdx.x;
    if (idx >= K * N) return;
    int n = idx / K, k = idx % K;
    Wt[(size_t)n * K + k] = bf16_cvt(W[(size_t)k * N + n]);
}

__global__ void pack_rowmajor_k(const float* __restrict__ S, int ldS,
                                unsigned short* __restrict__ Wt, int Nn, int K)
{
    int idx = blockIdx.x * blockDim.x + threadIdx.x;
    if (idx >= Nn * K) return;
    int n = idx / K, k = idx % K;
    Wt[(size_t)n * K + k] = bf16_cvt(S[(size_t)n * ldS + k]);
}

// ---------------------------------------------------------------------------
// LayerNorm over last dim (one wave per row)
// ---------------------------------------------------------------------------
__global__ void layernorm_k(const float* __restrict__ in, int ldi,
                            const float* __restrict__ g, const float* __restrict__ b,
                            float* __restrict__ out, int ldo, int M, int C)
{
    int row  = blockIdx.x * (blockDim.x >> 5) + (threadIdx.x >> 5);
    int lane = threadIdx.x & 31;
    if (row >= M) return;
    const float* p = in + (size_t)row * ldi;
    float s = 0.f, s2 = 0.f;
    for (int c = lane; c < C; c += 32) { float v = p[c]; s += v; s2 += v * v; }
#pragma unroll
    for (int o = 16; o > 0; o >>= 1) { s += __shfl_xor(s, o, 32); s2 += __shfl_xor(s2, o, 32); }
    float mean = s / C;
    float var  = s2 / C - mean * mean;
    float inv  = rsqrtf(var + 1e-5f);
    float* q = out + (size_t)row * ldo;
    for (int c = lane; c < C; c += 32) q[c] = (p[c] - mean) * inv * g[c] + b[c];
}

// ---------------------------------------------------------------------------
// Row softmax with scale (one wave per row, contiguous rows of length L)
// ---------------------------------------------------------------------------
__global__ void softmax_rows_k(float* __restrict__ X, long long rows, int L, float scale)
{
    long long row = (long long)blockIdx.x * (blockDim.x >> 5) + (threadIdx.x >> 5);
    int lane = threadIdx.x & 31;
    if (row >= rows) return;
    float* p = X + row * L;
    float mx = -3.4e38f;
    for (int i = lane; i < L; i += 32) mx = fmaxf(mx, p[i] * scale);
#pragma unroll
    for (int o = 16; o > 0; o >>= 1) mx = fmaxf(mx, __shfl_xor(mx, o, 32));
    float s = 0.f;
    for (int i = lane; i < L; i += 32) { float e = expf(p[i] * scale - mx); p[i] = e; s += e; }
#pragma unroll
    for (int o = 16; o > 0; o >>= 1) s += __shfl_xor(s, o, 32);
    float inv = 1.0f / s;
    for (int i = lane; i < L; i += 32) p[i] *= inv;
}

// ---------------------------------------------------------------------------
// Depthwise 3x3, NHWC, SAME; optional fused GELU on output
// ---------------------------------------------------------------------------
__global__ void dwconv3x3_k(const float* __restrict__ in, int ldIn,
                            const float* __restrict__ w, const float* __restrict__ b,
                            float* __restrict__ out, int ldOut,
                            int Bn, int H, int W, int C, int actGelu)
{
    long long idx = (long long)blockIdx.x * blockDim.x + threadIdx.x;
    long long total = (long long)Bn * H * W * C;
    if (idx >= total) return;
    int c = (int)(idx % C);
    long long p = idx / C;
    int x = (int)(p % W);
    long long t = p / W;
    int y  = (int)(t % H);
    int bb = (int)(t / H);
    float s = b ? b[c] : 0.0f;
#pragma unroll
    for (int dy = 0; dy < 3; ++dy) {
        int yy = y + dy - 1; if (yy < 0 || yy >= H) continue;
#pragma unroll
        for (int dx = 0; dx < 3; ++dx) {
            int xx = x + dx - 1; if (xx < 0 || xx >= W) continue;
            s += in[(((size_t)bb * H + yy) * W + xx) * (size_t)ldIn + c] * w[(dy * 3 + dx) * C + c];
        }
    }
    out[(size_t)p * ldOut + c] = actGelu ? gelu_f(s) : s;
}

// ---------------------------------------------------------------------------
// De/interleave along one axis: shape (outer, 2*len, inner)
// ---------------------------------------------------------------------------
__global__ void deinterleave_k(const float* __restrict__ in, float* __restrict__ ev,
                               float* __restrict__ od, long long outer, long long len, long long inner)
{
    long long idx = (long long)blockIdx.x * blockDim.x + threadIdx.x;
    long long total = outer * len * inner;
    if (idx >= total) return;
    long long i = idx % inner;
    long long t = idx / inner;
    long long l = t % len;
    long long o = t / len;
    const float* base = in + ((o * 2 * len) + 2 * l) * inner + i;
    ev[idx] = base[0];
    od[idx] = base[inner];
}

__global__ void interleave_k(const float* __restrict__ ev, const float* __restrict__ od,
                             float* __restrict__ out, long long outer, long long len, long long inner)
{
    long long idx = (long long)blockIdx.x * blockDim.x + threadIdx.x;
    long long total = outer * len * inner;
    if (idx >= total) return;
    long long i = idx % inner;
    long long t = idx / inner;
    long long l = t % len;
    long long o = t / len;
    float* base = out + ((o * 2 * len) + 2 * l) * inner + i;
    base[0]     = ev[idx];
    base[inner] = od[idx];
}

// out = a + sign*b  (per-row leading dims)
__global__ void combine_k(const float* __restrict__ a, int lda,
                          const float* __restrict__ b, int ldb,
                          float* __restrict__ o, int ldo,
                          long long rows, int C, float sign)
{
    long long idx = (long long)blockIdx.x * blockDim.x + threadIdx.x;
    if (idx >= rows * C) return;
    long long m = idx / C; int c = (int)(idx % C);
    o[m * ldo + c] = a[m * lda + c] + sign * b[m * ldb + c];
}

// hf = concat([h_ll, lh, h_hh], channel) -> [rows][384]
__global__ void concat3_k(const float* __restrict__ a, const float* __restrict__ b,
                          const float* __restrict__ c, float* __restrict__ o, long long rows)
{
    long long idx = (long long)blockIdx.x * blockDim.x + threadIdx.x;
    if (idx >= rows * 384) return;
    long long m = idx / 384; int ch = (int)(idx % 384);
    float v = (ch < 128) ? a[m * 128 + ch]
            : (ch < 256) ? b[m * 128 + (ch - 128)]
                         : c[m * 128 + (ch - 256)];
    o[idx] = v;
}

// Router final stage: logits = h2@W2 + b2 + eb ; softmax ; top-2 masked gates
__global__ void router_gate_k(const float* __restrict__ h2, const float* __restrict__ w2,
                              const float* __restrict__ b2, const float* __restrict__ eb,
                              float* __restrict__ gates, int M)
{
    int m = blockIdx.x * blockDim.x + threadIdx.x;
    if (m >= M) return;
    const float* r = h2 + (size_t)m * 96;
    float lg[4];
#pragma unroll
    for (int e = 0; e < 4; ++e) {
        float s = b2[e] + eb[e];
        for (int k = 0; k < 96; ++k) s += r[k] * w2[k * 4 + e];
        lg[e] = s;
    }
    float mx = fmaxf(fmaxf(lg[0], lg[1]), fmaxf(lg[2], lg[3]));
    float ex[4], ssum = 0.f;
#pragma unroll
    for (int e = 0; e < 4; ++e) { ex[e] = expf(lg[e] - mx); ssum += ex[e]; }
    int i0 = 0;
#pragma unroll
    for (int e = 1; e < 4; ++e) if (lg[e] > lg[i0]) i0 = e;
    int i1 = -1;
#pragma unroll
    for (int e = 0; e < 4; ++e) { if (e == i0) continue; if (i1 < 0 || lg[e] > lg[i1]) i1 = e; }
#pragma unroll
    for (int e = 0; e < 4; ++e)
        gates[(size_t)m * 4 + e] = (e == i0 || e == i1) ? (ex[e] / ssum) : 0.0f;
}

// per-pixel channel mean & max -> mm[p][2]
__global__ void chstats_k(const float* __restrict__ x, float* __restrict__ mm, long long M, int C)
{
    long long m = (long long)blockIdx.x * blockDim.x + threadIdx.x;
    if (m >= M) return;
    const float* p = x + m * C;
    float s = 0.f, mx = -3.4e38f;
    for (int c = 0; c < C; ++c) { float v = p[c]; s += v; mx = fmaxf(mx, v); }
    mm[m * 2 + 0] = s / C;
    mm[m * 2 + 1] = mx;
}

// 7x7 conv on 2-channel map -> sigmoid attention (no bias)
__global__ void saconv_k(const float* __restrict__ mm, const float* __restrict__ w,
                         float* __restrict__ att, int Bn, int H, int W)
{
    long long idx = (long long)blockIdx.x * blockDim.x + threadIdx.x;
    if (idx >= (long long)Bn * H * W) return;
    int x = (int)(idx % W);
    long long t = idx / W;
    int y  = (int)(t % H);
    int bb = (int)(t / H);
    float s = 0.f;
    for (int dy = 0; dy < 7; ++dy) {
        int yy = y + dy - 3; if (yy < 0 || yy >= H) continue;
        for (int dx = 0; dx < 7; ++dx) {
            int xx = x + dx - 3; if (xx < 0 || xx >= W) continue;
            const float* q = mm + (((size_t)bb * H + yy) * W + xx) * 2;
            s += q[0] * w[(dy * 7 + dx) * 2 + 0];
            s += q[1] * w[(dy * 7 + dx) * 2 + 1];
        }
    }
    att[idx] = sigmoid_f(s);
}

// y[m,c] += rs[c] * x[m,c] * (att ? att[m] : 1)
__global__ void residual_k(float* __restrict__ y, const float* __restrict__ x,
                           const float* __restrict__ rs, const float* __restrict__ att,
                           long long M, int C)
{
    long long idx = (long long)blockIdx.x * blockDim.x + threadIdx.x;
    if (idx >= M * C) return;
    long long m = idx / C; int c = (int)(idx % C);
    float v = rs[c] * x[idx];
    if (att) v *= att[m];
    y[idx] += v;
}

__global__ void gelu_copy_k(const float* __restrict__ in, float* __restrict__ out, long long n)
{
    long long i = (long long)blockIdx.x * blockDim.x + threadIdx.x;
    if (i < n) out[i] = gelu_f(in[i]);
}

__global__ void gelumul_k(const float* __restrict__ ac, const float* __restrict__ v,
                          float* __restrict__ o, long long n)
{
    long long i = (long long)blockIdx.x * blockDim.x + threadIdx.x;
    if (i < n) o[i] = gelu_f(ac[i]) * v[i];
}

// ---------------------------------------------------------------------------
// Host orchestration
// ---------------------------------------------------------------------------
namespace {

constexpr int Bn = 8, HH = 128, WW = 128, Cc = 128;
constexpr long long NPIX_FULL = (long long)Bn * HH * WW;   // 131072
constexpr long long NPIX_HALF = NPIX_FULL / 2;             // 65536
constexpr long long NPIX_QUAR = NPIX_FULL / 4;             // 32768
constexpr size_t FULL = (size_t)NPIX_FULL * Cc;            // 16,777,216 floats
constexpr size_t HALF = FULL / 2;
constexpr size_t QUAR = FULL / 4;
constexpr float  SCALE = 0.08838834764831845f;             // 128^-0.5

// ---- input indices (setup_inputs() insertion order, depth-first) ----
enum {
    I_X = 0, I_XT_W, I_XT_B, I_OT_W, I_OT_B,
    I_DWT = 5,    // 4 lifts x (dw_w, dw_b, pw_w, pw_b): P_horz, U_horz, P_vert, U_vert
    I_IDWT = 21,
    I_LNLOW_G = 37, I_LNLOW_B,
    I_QLOW_W = 39, I_QLOW_B, I_KLOW_W, I_KLOW_B,
    I_LNDL_G = 43, I_LNDL_B, I_DICT = 45,
    I_FC0_W = 46, I_FC0_B, I_FC1_W, I_FC1_B, I_FC2_W, I_FC2_B,
    I_EXP = 52, I_LNDH_G = 53, I_LNDH_B, I_LNH_G = 55, I_LNH_B,
    I_QH_W = 57, I_QH_B, I_KH_W, I_KH_B, I_VA_W, I_VA_B,
    I_EBIAS = 63,
    I_SW = 64, I_SB = 65, I_SAW = 66,
    I_DENSE = 67,  // 3 x (in_w, in_b, dw_w, dw_b, out_w, out_b)
    I_PROJ_W = 85, I_PROJ_B = 86,
    I_LNS_G = 87, I_LNS_B = 88, I_RS1 = 89,
    I_LNM_G = 90, I_LNM_B = 91,
    I_MLP_FC1W = 92, I_MLP_FC1B = 93, I_MLP_DWW = 94, I_MLP_DWB = 95,
    I_MLP_FC2W = 96, I_MLP_FC2B = 97, I_RS2 = 98
};

// ---- bf16 weight-pack arena offsets (ushort units) ----
enum : size_t {
    PK_XT   = 0,
    PK_PW   = PK_XT + 16384,            // 8 lifts x 128*128
    PK_QLOW = PK_PW + 8 * 16384,
    PK_KLOW = PK_QLOW + 16384,
    PK_DICT = PK_KLOW + 16384,          // 128 x 64
    PK_FC0  = PK_DICT + 8192,           // 384 x 512
    PK_FC1  = PK_FC0 + 196608,          // 96 x 384
    PK_QH   = PK_FC1 + 36864,           // 384 x 384
    PK_KH   = PK_QH + 147456,
    PK_VH   = PK_KH + 147456,
    PK_SW   = PK_VH + 147456,
    PK_DIN  = PK_SW + 16384,            // 3 x 128*128
    PK_DOUT = PK_DIN + 3 * 16384,
    PK_PROJ = PK_DOUT + 3 * 16384,      // 128 x 512
    PK_FC1M = PK_PROJ + 65536,          // 512 x 128
    PK_FC2  = PK_FC1M + 65536,          // 128 x 256
    PK_OUT  = PK_FC2 + 32768,
    PK_KD   = PK_OUT + 16384,           // 64 x 128  (runtime)
    PK_KEYS = PK_KD + 8192,             // 256 x 384 (runtime)
    PK_VALS = PK_KEYS + 98304,          // 4 x 384*64 (runtime)
    PK_END  = PK_VALS + 98304
};

// ---- small fp32 scratch offsets (float units) ----
enum : size_t {
    SM_LND  = 0,                        // 64 x 128
    SM_K    = SM_LND + 8192,            // 64 x 128
    SM_ED   = SM_K + 8192,              // 256 x 384
    SM_KEYS = SM_ED + 98304,
    SM_VALS = SM_KEYS + 98304,
    SM_MM   = SM_VALS + 98304,          // 131072 x 2
    SM_ATT  = SM_MM + 262144,           // 131072
    SM_END  = SM_ATT + 131072
};

inline long long cdiv(long long a, long long b) { return (a + b - 1) / b; }

} // namespace

extern "C" void kernel_launch(void* const* d_in, const int* in_sizes, int n_in,
                              void* d_out, int out_size, void* d_ws, size_t ws_size,
                              hipStream_t stream)
{
    (void)in_sizes; (void)n_in; (void)out_size; (void)ws_size;
    auto IN = [&](int i) -> const float* { return (const float*)d_in[i]; };

    char* wsb = (char*)d_ws;
    unsigned short* pk = (unsigned short*)wsb;                        //  8 MB pack arena
    float* sm = (float*)(wsb + (size_t)8 * 1024 * 1024);              // 16 MB smalls
    float* fb = (float*)(wsb + (size_t)24 * 1024 * 1024);             // 10 FULL slots
    auto F = [&](int i) -> float* { return fb + (size_t)i * FULL; };

    // ---------------- helpers ----------------
    auto gemm = [&](const float* A, int ldA, size_t wtOff, int ldW, const float* bias,
                    float* C, int ldC, long long M, int N, int K, int act,
                    const float* rs = nullptr, int rss = 0, int accf = 0) {
        dim3 g((unsigned)cdiv(N, 64), (unsigned)cdiv(M, 256));
        dim3 b(256);
        const unsigned short* W = pk + wtOff;
        if (N % 64 == 0) {
            if (act) gemm_wmma_bf16<1, true><<<g, b, 0, stream>>>(A, ldA, W, ldW, bias, C, ldC, (int)M, N, K, rs, rss, accf);
            else     gemm_wmma_bf16<0, true><<<g, b, 0, stream>>>(A, ldA, W, ldW, bias, C, ldC, (int)M, N, K, rs, rss, accf);
        } else {
            if (act) gemm_wmma_bf16<1, false><<<g, b, 0, stream>>>(A, ldA, W, ldW, bias, C, ldC, (int)M, N, K, rs, rss, accf);
            else     gemm_wmma_bf16<0, false><<<g, b, 0, stream>>>(A, ldA, W, ldW, bias, C, ldC, (int)M, N, K, rs, rss, accf);
        }
    };
    auto packT = [&](const float* W, size_t off, int K, int N) {
        pack_transpose_k<<<(unsigned)cdiv((long long)K * N, 256), 256, 0, stream>>>(W, pk + off, K, N);
    };
    auto packR = [&](const float* S, int ldS, size_t off, int Nn, int K) {
        pack_rowmajor_k<<<(unsigned)cdiv((long long)Nn * K, 256), 256, 0, stream>>>(S, ldS, pk + off, Nn, K);
    };
    auto layernorm = [&](const float* in, int ldi, const float* g, const float* b,
                         float* out, int ldo, long long M, int C) {
        layernorm_k<<<(unsigned)cdiv(M, 8), 256, 0, stream>>>(in, ldi, g, b, out, ldo, (int)M, C);
    };
    auto dwc = [&](const float* in, int ldIn, const float* w, const float* b, float* out, int ldOut,
                   int B_, int H_, int W_, int C_, int act) {
        long long n = (long long)B_ * H_ * W_ * C_;
        dwconv3x3_k<<<(unsigned)cdiv(n, 256), 256, 0, stream>>>(in, ldIn, w, b, out, ldOut, B_, H_, W_, C_, act);
    };
    auto combine = [&](const float* a, int lda, const float* b, int ldb, float* o, int ldo,
                       long long rows, int C, float sign) {
        combine_k<<<(unsigned)cdiv(rows * C, 256), 256, 0, stream>>>(a, lda, b, ldb, o, ldo, rows, C, sign);
    };
    // lift(x) = pw( gelu( dwconv3x3(x) ) ); paramBase -> dw_w,dw_b,pw_w,pw_b
    auto lift = [&](const float* in, int ldIn, int paramBase, size_t pwOff, float* tmp, float* out,
                    int B_, int H_, int W_) {
        dwc(in, ldIn, IN(paramBase + 0), IN(paramBase + 1), tmp, Cc, B_, H_, W_, Cc, 1);
        gemm(tmp, Cc, pwOff, Cc, IN(paramBase + 3), out, Cc, (long long)B_ * H_ * W_, Cc, Cc, 0);
    };

    // ---------------- pack all static weights (bf16, [N][K]) ----------------
    packT(IN(I_XT_W), PK_XT, 128, 128);
    for (int g = 0; g < 8; ++g) {
        int base = (g < 4) ? (I_DWT + g * 4) : (I_IDWT + (g - 4) * 4);
        packT(IN(base + 2), PK_PW + (size_t)g * 16384, 128, 128);
    }
    packT(IN(I_QLOW_W), PK_QLOW, 128, 128);
    packT(IN(I_KLOW_W), PK_KLOW, 128, 128);
    packT(IN(I_DICT),   PK_DICT, 64, 128);       // dict as [K=64][N=128]
    packT(IN(I_FC0_W),  PK_FC0, 512, 384);
    packT(IN(I_FC1_W),  PK_FC1, 384, 96);
    packT(IN(I_QH_W),   PK_QH, 384, 384);
    packT(IN(I_KH_W),   PK_KH, 384, 384);
    packT(IN(I_VA_W),   PK_VH, 384, 384);
    packT(IN(I_SW),     PK_SW, 128, 128);
    for (int i = 0; i < 3; ++i) {
        packT(IN(I_DENSE + i * 6 + 0), PK_DIN  + (size_t)i * 16384, 128, 128);
        packT(IN(I_DENSE + i * 6 + 4), PK_DOUT + (size_t)i * 16384, 128, 128);
    }
    packT(IN(I_PROJ_W),   PK_PROJ, 512, 128);
    packT(IN(I_MLP_FC1W), PK_FC1M, 128, 512);
    packT(IN(I_MLP_FC2W), PK_FC2, 256, 128);
    packT(IN(I_OT_W),     PK_OUT, 128, 128);

    // ---------------- 1) h = x @ x_trans ----------------
    gemm(IN(I_X), 128, PK_XT, 128, IN(I_XT_B), F(0), 128, NPIX_FULL, 128, 128, 0);

    // ---------------- 2) DWT (NHWC) ----------------
    float *evH = F(1), *odH = F(1) + HALF;              // half-size pair
    float *tH0 = F(2), *tH1 = F(2) + HALF;              // half lift temps
    float *hHz = F(3), *lHz = F(3) + HALF;              // h_horz / l_horz
    float *q0 = F(4), *q1 = F(4) + QUAR, *q2 = F(4) + 2 * QUAR, *q3 = F(4) + 3 * QUAR;
    float *hLL = F(5), *LH = F(5) + QUAR, *hHH = F(5) + 2 * QUAR, *LL = F(5) + 3 * QUAR;

    deinterleave_k<<<(unsigned)cdiv((long long)HALF, 256), 256, 0, stream>>>(F(0), evH, odH, 1024, 64, 128);
    lift(evH, 128, I_DWT + 0, PK_PW + 0 * 16384, tH0, tH1, Bn, 128, 64);     // P_horz
    combine(odH, 128, tH1, 128, hHz, 128, NPIX_HALF, 128, -1.0f);            // h_horz
    lift(hHz, 128, I_DWT + 4, PK_PW + 1 * 16384, tH0, tH1, Bn, 128, 64);     // U_horz
    combine(evH, 128, tH1, 128, lHz, 128, NPIX_HALF, 128, +1.0f);            // l_horz

    deinterleave_k<<<(unsigned)cdiv((long long)QUAR, 256), 256, 0, stream>>>(lHz, q0, q1, 8, 64, 8192);
    lift(q0, 128, I_DWT + 8, PK_PW + 2 * 16384, q2, q3, Bn, 64, 64);         // P_vert(even_ll)
    combine(q1, 128, q3, 128, hLL, 128, NPIX_QUAR, 128, -1.0f);              // h_ll
    lift(hLL, 128, I_DWT + 12, PK_PW + 3 * 16384, q2, q3, Bn, 64, 64);       // U_vert(h_ll)
    combine(q0, 128, q3, 128, LL, 128, NPIX_QUAR, 128, +1.0f);               // ll

    deinterleave_k<<<(unsigned)cdiv((long long)QUAR, 256), 256, 0, stream>>>(hHz, q0, q1, 8, 64, 8192);
    lift(q0, 128, I_DWT + 8, PK_PW + 2 * 16384, q2, q3, Bn, 64, 64);         // P_vert(even_hh)
    combine(q1, 128, q3, 128, hHH, 128, NPIX_QUAR, 128, -1.0f);              // h_hh
    lift(hHH, 128, I_DWT + 12, PK_PW + 3 * 16384, q2, q3, Bn, 64, 64);       // U_vert(h_hh)
    combine(q0, 128, q3, 128, LH, 128, NPIX_QUAR, 128, +1.0f);               // lh

    // ---------------- 3) low_freq ----------------
    float *llOut = F(6), *lnll = F(6) + QUAR, *qlow = F(6) + 2 * QUAR, *scoresL = F(6) + 3 * QUAR;
    layernorm(LL, 128, IN(I_LNLOW_G), IN(I_LNLOW_B), lnll, 128, NPIX_QUAR, 128);
    gemm(lnll, 128, PK_QLOW, 128, IN(I_QLOW_B), qlow, 128, NPIX_QUAR, 128, 128, 0);
    layernorm(IN(I_DICT), 128, IN(I_LNDL_G), IN(I_LNDL_B), sm + SM_LND, 128, 64, 128);
    gemm(sm + SM_LND, 128, PK_KLOW, 128, IN(I_KLOW_B), sm + SM_K, 128, 64, 128, 128, 0);
    packR(sm + SM_K, 128, PK_KD, 64, 128);                                   // k^T -> Wt[64][128]
    gemm(qlow, 128, PK_KD, 128, nullptr, scoresL, 64, NPIX_QUAR, 64, 128, 0);
    softmax_rows_k<<<(unsigned)cdiv(NPIX_QUAR, 8), 256, 0, stream>>>(scoresL, NPIX_QUAR, 64, SCALE);
    gemm(scoresL, 64, PK_DICT, 64, nullptr, llOut, 128, NPIX_QUAR, 128, 64, 0);

    // ---------------- 4) high_freq ----------------
    float *hf = F(7), *hfln = F(8), *qh = F(9), *attn = F(2);                // attn [32768][256]
    concat3_k<<<(unsigned)cdiv(NPIX_QUAR * 384, 256), 256, 0, stream>>>(hLL, LH, hHH, hf, NPIX_QUAR);
    layernorm(hf, 384, IN(I_LNH_G), IN(I_LNH_B), hfln, 384, NPIX_QUAR, 384);
    gemm(hfln, 384, PK_QH, 384, IN(I_QH_B), qh, 384, NPIX_QUAR, 384, 384, 0);
    layernorm(IN(I_EXP), 384, IN(I_LNDH_G), IN(I_LNDH_B), sm + SM_ED, 384, 256, 384);
    gemm(sm + SM_ED, 384, PK_KH, 384, IN(I_KH_B), sm + SM_KEYS, 384, 256, 384, 384, 0);
    gemm(sm + SM_ED, 384, PK_VH, 384, IN(I_VA_B), sm + SM_VALS, 384, 256, 384, 384, 0);
    packR(sm + SM_KEYS, 384, PK_KEYS, 256, 384);                             // keys^T blocks
    for (int e = 0; e < 4; ++e)                                              // vals blocks [384][64]
        packT(sm + SM_VALS + (size_t)e * 64 * 384, PK_VALS + (size_t)e * 384 * 64, 64, 384);
    for (int e = 0; e < 4; ++e)                                              // scores per expert
        gemm(qh, 384, PK_KEYS + (size_t)e * 64 * 384, 384, nullptr,
             attn + e * 64, 256, NPIX_QUAR, 64, 384, 0);
    softmax_rows_k<<<(unsigned)cdiv(NPIX_QUAR * 4, 8), 256, 0, stream>>>(attn, NPIX_QUAR * 4, 64, SCALE);

    // router (split-K concat: hf part + lf part), gates
    float *h1 = F(8), *h2 = F(6) + QUAR, *gates = F(6) + 2 * QUAR;           // reuse dead slots
    gemm(hf, 384, PK_FC0, 512, nullptr, h1, 384, NPIX_QUAR, 384, 384, 0);
    gemm(llOut, 128, PK_FC0 + 384, 512, IN(I_FC0_B), h1, 384, NPIX_QUAR, 384, 128, 1,
         nullptr, 0, 1);                                                     // gelu after accumulate
    gemm(h1, 384, PK_FC1, 384, IN(I_FC1_B), h2, 96, NPIX_QUAR, 96, 384, 1);
    router_gate_k<<<(unsigned)cdiv(NPIX_QUAR, 256), 256, 0, stream>>>(
        h2, IN(I_FC2_W), IN(I_FC2_B), IN(I_EBIAS), gates, (int)NPIX_QUAR);

    // hf_out = sum_e gates[:,e] * (attn_e @ vals_e)   (gated accumulate epilogue)
    float* hfOut = F(9);                                                     // qh dead
    for (int e = 0; e < 4; ++e)
        gemm(attn + e * 64, 256, PK_VALS + (size_t)e * 384 * 64, 64, nullptr,
             hfOut, 384, NPIX_QUAR, 384, 64, 0, gates + e, 4, (e > 0));

    // ---------------- 5) IDWT ----------------
    const float *pLH = hfOut, *pHL = hfOut + 128, *pHH2 = hfOut + 256;       // ld 384
    // l_horz' = inv_lift(ll_out, lh, P_vert, U_vert, H)
    lift(pLH, 384, I_IDWT + 12, PK_PW + 7 * 16384, q2, q3, Bn, 64, 64);      // U_vert(lh)
    combine(llOut, 128, q3, 128, q0, 128, NPIX_QUAR, 128, -1.0f);            // even
    lift(q0, 128, I_IDWT + 8, PK_PW + 6 * 16384, q2, q3, Bn, 64, 64);        // P_vert(even)
    combine(pLH, 384, q3, 128, q1, 128, NPIX_QUAR, 128, +1.0f);              // odd
    interleave_k<<<(unsigned)cdiv((long long)QUAR, 256), 256, 0, stream>>>(q0, q1, lHz, 8, 64, 8192);
    // h_horz' = inv_lift(hl, hh, P_vert, U_vert, H)
    lift(pHH2, 384, I_IDWT + 12, PK_PW + 7 * 16384, q2, q3, Bn, 64, 64);
    combine(pHL, 384, q3, 128, q0, 128, NPIX_QUAR, 128, -1.0f);
    lift(q0, 128, I_IDWT + 8, PK_PW + 6 * 16384, q2, q3, Bn, 64, 64);
    combine(pHH2, 384, q3, 128, q1, 128, NPIX_QUAR, 128, +1.0f);
    interleave_k<<<(unsigned)cdiv((long long)QUAR, 256), 256, 0, stream>>>(q0, q1, hHz, 8, 64, 8192);
    // rec = inv_lift(l_horz', h_horz', P_horz, U_horz, W)
    lift(hHz, 128, I_IDWT + 4, PK_PW + 5 * 16384, tH0, tH1, Bn, 128, 64);    // U_horz(high)
    combine(lHz, 128, tH1, 128, evH, 128, NPIX_HALF, 128, -1.0f);
    lift(evH, 128, I_IDWT + 0, PK_PW + 4 * 16384, tH0, tH1, Bn, 128, 64);    // P_horz(even)
    combine(hHz, 128, tH1, 128, odH, 128, NPIX_HALF, 128, +1.0f);
    float* y = F(0);
    interleave_k<<<(unsigned)cdiv((long long)HALF, 256), 256, 0, stream>>>(evH, odH, y, 1024, 64, 128);

    // ---------------- 6) MSA ----------------
    float *yln = F(1), *sbuf = F(5), *proj = F(3);
    layernorm(y, 128, IN(I_LNS_G), IN(I_LNS_B), yln, 128, NPIX_FULL, 128);
    gemm(yln, 128, PK_SW, 128, IN(I_SB), sbuf, 128, NPIX_FULL, 128, 128, 0);
    gemm(sbuf, 128, PK_PROJ + 0 * 128, 512, nullptr, proj, 128, NPIX_FULL, 128, 128, 0);
    const float* prev = sbuf;
    for (int i = 0; i < 3; ++i) {
        float *g1 = F(8), *tmid = F(7), *t2 = F(8);
        float* outBuf = (i % 2 == 0) ? F(6) : F(9);
        int pb = I_DENSE + i * 6;
        gelu_copy_k<<<(unsigned)cdiv((long long)FULL, 256), 256, 0, stream>>>(prev, g1, (long long)FULL);
        gemm(g1, 128, PK_DIN + (size_t)i * 16384, 128, IN(pb + 1), tmid, 128, NPIX_FULL, 128, 128, 1);
        dwc(tmid, 128, IN(pb + 2), IN(pb + 3), t2, 128, Bn, HH, WW, 128, 1);
        gemm(t2, 128, PK_DOUT + (size_t)i * 16384, 128, IN(pb + 5), outBuf, 128, NPIX_FULL, 128, 128, 0);
        gemm(outBuf, 128, PK_PROJ + (size_t)(i + 1) * 128, 512,
             (i == 2) ? IN(I_PROJ_B) : nullptr, proj, 128, NPIX_FULL, 128, 128, 0, nullptr, 0, 1);
        prev = outBuf;
    }
    chstats_k<<<(unsigned)cdiv(NPIX_FULL, 256), 256, 0, stream>>>(proj, sm + SM_MM, NPIX_FULL, 128);
    saconv_k<<<(unsigned)cdiv(NPIX_FULL, 256), 256, 0, stream>>>(sm + SM_MM, IN(I_SAW), sm + SM_ATT, Bn, HH, WW);
    residual_k<<<(unsigned)cdiv((long long)FULL, 256), 256, 0, stream>>>(
        y, proj, IN(I_RS1), sm + SM_ATT, NPIX_FULL, 128);

    // ---------------- 7) MLP ----------------
    float *yln2 = F(1), *aBuf = F(4), *vBuf = F(6), *acBuf = F(8), *mlpOut = F(2);
    layernorm(y, 128, IN(I_LNM_G), IN(I_LNM_B), yln2, 128, NPIX_FULL, 128);
    gemm(yln2, 128, PK_FC1M, 128, IN(I_MLP_FC1B), aBuf, 256, NPIX_FULL, 256, 128, 0);
    gemm(yln2, 128, PK_FC1M + (size_t)256 * 128, 128, IN(I_MLP_FC1B) + 256, vBuf, 256,
         NPIX_FULL, 256, 128, 0);
    dwc(aBuf, 256, IN(I_MLP_DWW), IN(I_MLP_DWB), acBuf, 256, Bn, HH, WW, 256, 0);
    gelumul_k<<<(unsigned)cdiv(NPIX_FULL * 256, 256), 256, 0, stream>>>(
        acBuf, vBuf, aBuf, NPIX_FULL * 256);
    gemm(aBuf, 256, PK_FC2, 256, IN(I_MLP_FC2B), mlpOut, 128, NPIX_FULL, 128, 256, 0);
    residual_k<<<(unsigned)cdiv((long long)FULL, 256), 256, 0, stream>>>(
        y, mlpOut, IN(I_RS2), nullptr, NPIX_FULL, 128);

    // ---------------- 8) output ----------------
    gemm(y, 128, PK_OUT, 128, IN(I_OT_B), (float*)d_out, 128, NPIX_FULL, 128, 128, 0);
}
